// WaveNetModel_53154515255600
// MI455X (gfx1250) — matrix operand order repaired and checked by reading, not compiled, not run on hardware
//
#include <hip/hip_runtime.h>
#include <hip/hip_bf16.h>

typedef _Float16 half16 __attribute__((ext_vector_type(16)));
typedef _Float16 half8  __attribute__((ext_vector_type(8)));
typedef float    float8 __attribute__((ext_vector_type(8)));
typedef float    float4v __attribute__((ext_vector_type(4)));

#define BATCH 8
#define TLEN  16384
#define CH    32
#define TILES_PER_B (TLEN / 16)          // 1024 (16-wide tiles: input/output kernels)
#define NTILES (BATCH * TILES_PER_B)     // 8192 waves
#define TILES32_PER_B (TLEN / 32)        // 512  (32-wide tiles: layer kernel)
#define NTILES32 (BATCH * TILES32_PER_B) // 4096 waves
#define NFRAG 514                        // 16 (in) + 480 (layers) + 2 (s1) + 16 (s2)

// ---------------------------------------------------------------------------
// WMMA helper: D = A(16x32 f16) * B(32x16 f16) + C(16x16 f32)
// ---------------------------------------------------------------------------
__device__ __forceinline__ float8 wmma_f16(half16 a, half16 b, float8 c) {
  return __builtin_amdgcn_wmma_f32_16x16x32_f16(
      /*neg_a=*/false, a, /*neg_b=*/false, b,
      /*c_mod=*/(short)0, c, /*reuse_a=*/false, /*reuse_b=*/false);
}

// Load one pre-swizzled A fragment (32 lanes x 16 halves, 1KB per fragment).
__device__ __forceinline__ half16 load_frag(const _Float16* wpack, int f, int lane) {
  return *(const half16*)(wpack + (size_t)f * 512 + lane * 16);
}

// B fragment from h[B][T][32] f16, time column t (already includes lane row).
// B layout: lane L -> col N = L&15, elements 0..15 -> K = (L<16 ? e : 16+e),
// K maps to channel index (contiguous in memory). t<0 -> zeros (causal pad).
__device__ __forceinline__ half16 load_h_bfrag(const _Float16* h, int b, int t, int lane) {
  if (t < 0) { half16 z = {}; return z; }
  return *(const half16*)(h + ((size_t)b * TLEN + t) * CH + ((lane >> 4) << 4));
}

// ---------------------------------------------------------------------------
// Weight pre-swizzle: build A fragments in contiguous-K layout (same K order as the B fragments).
// A layout: lane L: M = L&15; element e:
//   lane<16 : K = e        -> K in {0..15}
//   lane>=16: K = e + 16   -> K in {16..31}
// Fragment map: [0..15] in_w ([mt][kc], mt:2, kc:8)
//               [16..495] per layer 12: conv[mt][kc]x4, gate x4, res[mt]x2, skip[mt]x2
//               [496..497] s1, [498..513] s2 (16 M tiles)
// ---------------------------------------------------------------------------
__global__ void pack_weights(const float* __restrict__ in_w,
                             const float* __restrict__ conv_w,
                             const float* __restrict__ gate_w,
                             const float* __restrict__ res_w,
                             const float* __restrict__ skip_w,
                             const float* __restrict__ s1_w,
                             const float* __restrict__ s2_w,
                             _Float16* __restrict__ wpack) {
  int tid = blockIdx.x * blockDim.x + threadIdx.x;
  if (tid >= NFRAG * 32) return;
  int f = tid >> 5, lane = tid & 31;
  int row = lane & 15;
  half16 out;
  for (int e = 0; e < 16; ++e) {
    int kl = (lane < 16) ? e : e + 16;
    float v;
    if (f < 16) {                       // in_w [32,256]
      int mt = f >> 3, kc = f & 7;
      v = in_w[(mt * 16 + row) * 256 + kc * 32 + kl];
    } else if (f < 496) {               // per-layer
      int g = f - 16, layer = g / 12, j = g % 12;
      if (j < 8) {                      // conv / gate [40,32,32,2], K = i + 32*k
        const float* w = (j < 4) ? conv_w : gate_w;
        int jj = j & 3, mt = jj >> 1, kc = jj & 1;
        v = w[((layer * 32 + mt * 16 + row) * 32 + kl) * 2 + kc];
      } else {                          // res / skip [40,32,32,1]
        const float* w = (j < 10) ? res_w : skip_w;
        int mt = j & 1;
        v = w[(layer * 32 + mt * 16 + row) * 32 + kl];
      }
    } else if (f < 498) {               // s1 [32,32]
      int mt = f - 496;
      v = s1_w[(mt * 16 + row) * 32 + kl];
    } else {                            // s2 [256,32]
      int mt = f - 498;
      v = s2_w[(mt * 16 + row) * 32 + kl];
    }
    out[e] = (_Float16)v;
  }
  *(half16*)(wpack + (size_t)f * 512 + lane * 16) = out;
}

// ---------------------------------------------------------------------------
// Input 1x1 conv: 256 -> 32.  GEMM M=32, K=256 (8 chunks), N=16 per wave.
// x is [B,256,T] (channel-major): for fixed channel, lanes 0..15 read 16
// consecutive floats -> coalesced. One-shot 128MB read.
// ---------------------------------------------------------------------------
__global__ __launch_bounds__(256) void input_pw(const float* __restrict__ x,
                                                const float* __restrict__ in_b,
                                                const _Float16* __restrict__ wpack,
                                                _Float16* __restrict__ h0) {
  int wave = (blockIdx.x * blockDim.x + threadIdx.x) >> 5;
  int lane = threadIdx.x & 31;
  int b = wave >> 10;
  int t0 = (wave & (TILES_PER_B - 1)) << 4;
  int row = lane & 15, hi = lane >> 4;
  int tcol = t0 + row;

  float8 c0, c1;
  for (int r = 0; r < 8; ++r) {
    c0[r] = in_b[r + hi * 8];
    c1[r] = in_b[16 + r + hi * 8];
  }
  for (int kc = 0; kc < 8; ++kc) {
    half16 bf;
    const float* xp = x + ((size_t)b * 256 + kc * 32 + hi * 16) * TLEN + tcol;
    for (int e = 0; e < 16; ++e) bf[e] = (_Float16)xp[(size_t)e * TLEN];
    c0 = wmma_f16(load_frag(wpack, kc, lane), bf, c0);
    c1 = wmma_f16(load_frag(wpack, 8 + kc, lane), bf, c1);
  }
  size_t base = ((size_t)b * TLEN + tcol) * CH;
  half8 n0, n1;
  for (int r = 0; r < 8; ++r) { n0[r] = (_Float16)c0[r]; n1[r] = (_Float16)c1[r]; }
  half8* hp = (half8*)(h0 + base);
  hp[hi] = n0; hp[2 + hi] = n1;
}

// ---------------------------------------------------------------------------
// One WaveNet layer (launched 40x, h ping-pong).  Each wave owns 32 time
// steps = two 16-column N tiles sharing every A (weight) fragment:
//   24 WMMAs per wave, weight fragments fetched once.
// ---------------------------------------------------------------------------
__global__ __launch_bounds__(256) void wavenet_layer(const _Float16* __restrict__ hin,
                                                     _Float16* __restrict__ hout,
                                                     float* __restrict__ skip,
                                                     const _Float16* __restrict__ wpack,
                                                     const float* __restrict__ conv_b,
                                                     const float* __restrict__ gate_b,
                                                     const float* __restrict__ res_b,
                                                     const float* __restrict__ skip_b,
                                                     int layer, int dil, int first) {
  int wave = (blockIdx.x * blockDim.x + threadIdx.x) >> 5;   // 0..4095
  int lane = threadIdx.x & 31;
  int b = wave >> 9;                         // TILES32_PER_B = 512
  int t0 = (wave & (TILES32_PER_B - 1)) << 5;
  int row = lane & 15, hi = lane >> 4;
  int ta = t0 + row;        // N tile 0 column
  int tb = ta + 16;         // N tile 1 column
  int fbase = 16 + layer * 12;

  // bias init for both N tiles (same values)
  const float* cb = conv_b + layer * 32;
  const float* gb = gate_b + layer * 32;
  float8 c0a, c1a, g0a, g1a, c0b, c1b, g0b, g1b;
  for (int r = 0; r < 8; ++r) {
    int m = r + hi * 8;
    float vc0 = cb[m], vc1 = cb[16 + m], vg0 = gb[m], vg1 = gb[16 + m];
    c0a[r] = vc0; c0b[r] = vc0;  c1a[r] = vc1; c1b[r] = vc1;
    g0a[r] = vg0; g0b[r] = vg0;  g1a[r] = vg1; g1b[r] = vg1;
  }

  // B fragments: K chunk 0 = tap at t-d, chunk 1 = tap at t; two N tiles
  half16 b0a = load_h_bfrag(hin, b, ta - dil, lane);
  half16 b1a = load_h_bfrag(hin, b, ta, lane);
  half16 b0b = load_h_bfrag(hin, b, tb - dil, lane);
  half16 b1b = load_h_bfrag(hin, b, tb, lane);

  half16 A;
  A = load_frag(wpack, fbase + 0, lane); c0a = wmma_f16(A, b0a, c0a); c0b = wmma_f16(A, b0b, c0b);
  A = load_frag(wpack, fbase + 1, lane); c0a = wmma_f16(A, b1a, c0a); c0b = wmma_f16(A, b1b, c0b);
  A = load_frag(wpack, fbase + 2, lane); c1a = wmma_f16(A, b0a, c1a); c1b = wmma_f16(A, b0b, c1b);
  A = load_frag(wpack, fbase + 3, lane); c1a = wmma_f16(A, b1a, c1a); c1b = wmma_f16(A, b1b, c1b);
  A = load_frag(wpack, fbase + 4, lane); g0a = wmma_f16(A, b0a, g0a); g0b = wmma_f16(A, b0b, g0b);
  A = load_frag(wpack, fbase + 5, lane); g0a = wmma_f16(A, b1a, g0a); g0b = wmma_f16(A, b1b, g0b);
  A = load_frag(wpack, fbase + 6, lane); g1a = wmma_f16(A, b0a, g1a); g1b = wmma_f16(A, b0b, g1b);
  A = load_frag(wpack, fbase + 7, lane); g1a = wmma_f16(A, b1a, g1a); g1b = wmma_f16(A, b1b, g1b);

  // gating
  float8 a0a, a1a, a0b, a1b;
  for (int r = 0; r < 8; ++r) {
    a0a[r] = tanhf(c0a[r]) * (1.0f / (1.0f + __expf(-g0a[r])));
    a1a[r] = tanhf(c1a[r]) * (1.0f / (1.0f + __expf(-g1a[r])));
    a0b[r] = tanhf(c0b[r]) * (1.0f / (1.0f + __expf(-g0b[r])));
    a1b[r] = tanhf(c1b[r]) * (1.0f / (1.0f + __expf(-g1b[r])));
  }

  // Rearrange C-layout activations (lane=col, 8 rows) into B fragments:
  // lane<16 needs channels 0..15 of its column (rows 8..15 live in lane^16),
  // lane>=16 needs channels 16..31.
  half16 afa, afb;
  for (int r = 0; r < 8; ++r) {
    float p0a = __shfl_xor(a0a[r], 16, 32);
    float p1a = __shfl_xor(a1a[r], 16, 32);
    float p0b = __shfl_xor(a0b[r], 16, 32);
    float p1b = __shfl_xor(a1b[r], 16, 32);
    if (hi == 0) {
      afa[r] = (_Float16)a0a[r]; afa[r + 8] = (_Float16)p0a;
      afb[r] = (_Float16)a0b[r]; afb[r + 8] = (_Float16)p0b;
    } else {
      afa[r] = (_Float16)p1a;    afa[r + 8] = (_Float16)a1a[r];
      afb[r] = (_Float16)p1b;    afb[r + 8] = (_Float16)a1b[r];
    }
  }

  const float* rb = res_b + layer * 32;
  const float* sb = skip_b + layer * 32;
  float8 r0a, r1a, s0a, s1a, r0b, r1b, s0b, s1b;
  for (int r = 0; r < 8; ++r) {
    int m = r + hi * 8;
    float vr0 = rb[m], vr1 = rb[16 + m], vs0 = sb[m], vs1 = sb[16 + m];
    r0a[r] = vr0; r0b[r] = vr0;  r1a[r] = vr1; r1b[r] = vr1;
    s0a[r] = vs0; s0b[r] = vs0;  s1a[r] = vs1; s1b[r] = vs1;
  }
  A = load_frag(wpack, fbase +  8, lane); r0a = wmma_f16(A, afa, r0a); r0b = wmma_f16(A, afb, r0b);
  A = load_frag(wpack, fbase +  9, lane); r1a = wmma_f16(A, afa, r1a); r1b = wmma_f16(A, afb, r1b);
  A = load_frag(wpack, fbase + 10, lane); s0a = wmma_f16(A, afa, s0a); s0b = wmma_f16(A, afb, s0b);
  A = load_frag(wpack, fbase + 11, lane); s1a = wmma_f16(A, afa, s1a); s1b = wmma_f16(A, afb, s1b);

  // residual update + skip accumulation for both N tiles
  for (int n = 0; n < 2; ++n) {
    int t = (n == 0) ? ta : tb;
    float8 r0 = (n == 0) ? r0a : r0b;
    float8 r1 = (n == 0) ? r1a : r1b;
    float8 s0 = (n == 0) ? s0a : s0b;
    float8 s1 = (n == 0) ? s1a : s1b;

    size_t base = ((size_t)b * TLEN + t) * CH;
    const half8* hop = (const half8*)(hin + base);
    half8 old0 = hop[hi], old1 = hop[2 + hi];
    half8 n0, n1;
    for (int r = 0; r < 8; ++r) {
      n0[r] = (_Float16)(r0[r] + (float)old0[r]);
      n1[r] = (_Float16)(r1[r] + (float)old1[r]);
    }
    half8* hnp = (half8*)(hout + base);
    hnp[hi] = n0; hnp[2 + hi] = n1;

    // skip: vector f32 RMW (layer 0 stores -> no pre-zero needed)
    float4v* q0 = (float4v*)(skip + base + hi * 8);
    float4v* q1 = (float4v*)(skip + base + 16 + hi * 8);
    float4v u00 = {s0[0], s0[1], s0[2], s0[3]};
    float4v u01 = {s0[4], s0[5], s0[6], s0[7]};
    float4v u10 = {s1[0], s1[1], s1[2], s1[3]};
    float4v u11 = {s1[4], s1[5], s1[6], s1[7]};
    if (!first) {
      u00 += q0[0]; u01 += q0[1];
      u10 += q1[0]; u11 += q1[1];
    }
    q0[0] = u00; q0[1] = u01;
    q1[0] = u10; q1[1] = u11;
  }
}

// ---------------------------------------------------------------------------
// Output head: out = s2 * relu(s1 * skip + s1_b) + s2_b   (M=256)
// ---------------------------------------------------------------------------
__global__ __launch_bounds__(256) void output_pw(const float* __restrict__ skip,
                                                 const _Float16* __restrict__ wpack,
                                                 const float* __restrict__ s1_b,
                                                 const float* __restrict__ s2_b,
                                                 float* __restrict__ out) {
  int wave = (blockIdx.x * blockDim.x + threadIdx.x) >> 5;
  int lane = threadIdx.x & 31;
  int b = wave >> 10;
  int t0 = (wave & (TILES_PER_B - 1)) << 4;
  int row = lane & 15, hi = lane >> 4;
  int tcol = t0 + row;

  // B fragment from skip (f32 -> f16), channels contiguous
  const float* sp = skip + ((size_t)b * TLEN + tcol) * CH + hi * 16;
  half16 bf;
  for (int e = 0; e < 16; ++e) bf[e] = (_Float16)sp[e];

  float8 c0, c1;
  for (int r = 0; r < 8; ++r) {
    c0[r] = s1_b[r + hi * 8];
    c1[r] = s1_b[16 + r + hi * 8];
  }
  c0 = wmma_f16(load_frag(wpack, 496, lane), bf, c0);
  c1 = wmma_f16(load_frag(wpack, 497, lane), bf, c1);
  for (int r = 0; r < 8; ++r) {
    c0[r] = fmaxf(c0[r], 0.0f);
    c1[r] = fmaxf(c1[r], 0.0f);
  }

  // repack to B fragment
  half16 af;
  for (int r = 0; r < 8; ++r) {
    float p0 = __shfl_xor(c0[r], 16, 32);
    float p1 = __shfl_xor(c1[r], 16, 32);
    if (hi == 0) { af[r] = (_Float16)c0[r]; af[r + 8] = (_Float16)p0; }
    else         { af[r] = (_Float16)p1;    af[r + 8] = (_Float16)c1[r]; }
  }

  // s2: 16 M tiles of 16 output channels
  for (int mt = 0; mt < 16; ++mt) {
    float8 o;
    for (int r = 0; r < 8; ++r) o[r] = s2_b[mt * 16 + hi * 8 + r];
    o = wmma_f16(load_frag(wpack, 498 + mt, lane), af, o);
    float* op = out + ((size_t)b * 256 + mt * 16 + hi * 8) * TLEN + tcol;
    for (int r = 0; r < 8; ++r) op[(size_t)r * TLEN] = o[r];
  }
}

// ---------------------------------------------------------------------------
extern "C" void kernel_launch(void* const* d_in, const int* in_sizes, int n_in,
                              void* d_out, int out_size, void* d_ws, size_t ws_size,
                              hipStream_t stream) {
  const float* x      = (const float*)d_in[0];
  const float* in_w   = (const float*)d_in[1];
  const float* in_b   = (const float*)d_in[2];
  const float* conv_w = (const float*)d_in[3];
  const float* conv_b = (const float*)d_in[4];
  const float* gate_w = (const float*)d_in[5];
  const float* gate_b = (const float*)d_in[6];
  const float* res_w  = (const float*)d_in[7];
  const float* res_b  = (const float*)d_in[8];
  const float* skip_w = (const float*)d_in[9];
  const float* skip_b = (const float*)d_in[10];
  const float* s1_w   = (const float*)d_in[11];
  const float* s1_b   = (const float*)d_in[12];
  const float* s2_w   = (const float*)d_in[13];
  const float* s2_b   = (const float*)d_in[14];

  const size_t HBYTES = (size_t)BATCH * TLEN * CH * sizeof(_Float16); // 8 MB
  _Float16* wpack = (_Float16*)d_ws;
  _Float16* h0    = (_Float16*)((char*)d_ws + (1u << 20));
  _Float16* h1    = (_Float16*)((char*)d_ws + (1u << 20) + HBYTES);
  float*    skip  = (float*)((char*)d_ws + (1u << 20) + 2 * HBYTES);

  pack_weights<<<(NFRAG * 32 + 255) / 256, 256, 0, stream>>>(
      in_w, conv_w, gate_w, res_w, skip_w, s1_w, s2_w, wpack);

  input_pw<<<NTILES / 8, 256, 0, stream>>>(x, in_b, wpack, h0);

  _Float16* cur = h0;
  _Float16* nxt = h1;
  for (int i = 0; i < 40; ++i) {
    int d = 1 << (i % 10);
    wavenet_layer<<<NTILES32 / 8, 256, 0, stream>>>(cur, nxt, skip, wpack,
                                                    conv_b, gate_b, res_b, skip_b,
                                                    i, d, (i == 0) ? 1 : 0);
    _Float16* t = cur; cur = nxt; nxt = t;
  }

  output_pw<<<NTILES / 8, 256, 0, stream>>>(skip, wpack, s1_b, s2_b, (float*)d_out);
}
